// Postfilter_21964462751930
// MI455X (gfx1250) — 8-bit matrix operand order repaired and checked by reading, compiled, not run on hardware
//
#include <hip/hip_runtime.h>
#include <hip/hip_bf16.h>
#include <stdint.h>

#define NDET   128
#define IMG_H  608
#define IMG_W  1088
#define NPIX   (IMG_H * IMG_W)
#define CHUNK  1024
#define NCHUNK (NPIX / CHUNK)
#define STRIDE (CHUNK + 16)

static_assert(NPIX % CHUNK == 0, "chunking must be exact");
static_assert(STRIDE % 16 == 0, "need 16B-aligned LDS rows for b128 loads");

typedef __attribute__((ext_vector_type(8))) int v8i;

__global__ void pf_zero_kernel(int* __restrict__ p, int n) {
    int i = blockIdx.x * blockDim.x + threadIdx.x;
    if (i < n) p[i] = 0;
}

__global__ void pf_gram_kernel(const float* __restrict__ masks,
                               int* __restrict__ inter) {
    extern __shared__ __align__(16) uint8_t sm[];

    const int t    = threadIdx.x;
    const int wave = t >> 5;
    const int lane = t & 31;
    const int half = lane >> 4;
    const int l16  = lane & 15;

    v8i acc[8];
    #pragma unroll
    for (int c = 0; c < 8; ++c) acc[c] = (v8i)0;

    const int h8  = half * 8;
    const int h16 = half * 16;
    uint8_t* aRow = sm + (size_t)(wave * 16 + l16) * STRIDE;

    for (int chunk = blockIdx.x; chunk < NCHUNK; chunk += gridDim.x) {
        const float4* src = (const float4*)(masks + (size_t)chunk * CHUNK);
        const size_t rowStride4 = NPIX / 4;
        #pragma unroll 4
        for (int row = 0; row < NDET; ++row) {
            float4 f = src[(size_t)row * rowStride4 + t];
            uint32_t p = (f.x > 0.5f ? 0x01u       : 0u)
                       | (f.y > 0.5f ? 0x0100u     : 0u)
                       | (f.z > 0.5f ? 0x010000u   : 0u)
                       | (f.w > 0.5f ? 0x01000000u : 0u);
            *(uint32_t*)(sm + (size_t)row * STRIDE + t * 4) = p;
        }
        __syncthreads();

        for (int s = 0; s < CHUNK / 64; ++s) {
            const int kb = s * 64;
            uint2 q0 = *(const uint2*)(aRow + kb + 0  + h16);
            uint2 q1 = *(const uint2*)(aRow + kb + 8  + h16);
            uint2 q2 = *(const uint2*)(aRow + kb + 32 + h16);
            uint2 q3 = *(const uint2*)(aRow + kb + 40 + h16);
            v8i a = { (int)q0.x, (int)q0.y, (int)q1.x, (int)q1.y,
                      (int)q2.x, (int)q2.y, (int)q3.x, (int)q3.y };
            #pragma unroll
            for (int c = 0; c < 8; ++c) {
                const uint8_t* bRow =
                    sm + (size_t)(c * 16 + l16) * STRIDE + kb + h16;
                uint4 b0 = *(const uint4*)(bRow);
                uint4 b1 = *(const uint4*)(bRow + 32);
                v8i b = { (int)b0.x, (int)b0.y, (int)b0.z, (int)b0.w,
                          (int)b1.x, (int)b1.y, (int)b1.z, (int)b1.w };
                acc[c] = __builtin_amdgcn_wmma_i32_16x16x64_iu8(
                     false, a,  false, b, acc[c],
                     false,  false);
            }
        }
        __syncthreads();
    }

    #pragma unroll
    for (int c = 0; c < 8; ++c) {
        const int n = c * 16 + l16;
        #pragma unroll
        for (int g = 0; g < 8; ++g) {
            const int m = wave * 16 + g + half * 8;
            atomicAdd(inter + m * NDET + n, acc[c][g]);
        }
    }
}

__global__ void pf_epilogue_kernel(const float* __restrict__ boxes,
                                   const float* __restrict__ scores,
                                   const int*   __restrict__ labels,
                                   const int*   __restrict__ inter,
                                   float*       __restrict__ out) {
    __shared__ float sScore[NDET], sArea[NDET];
    __shared__ float sX1[NDET], sY1[NDET], sX2[NDET], sY2[NDET];
    __shared__ int   sOrder[NDET], sKeepS[NDET], sKeepO[NDET], sLabel[NDET];

    const int t = threadIdx.x;
    sScore[t] = scores[t];
    sLabel[t] = labels[t];
    sX1[t] = boxes[t * 4 + 0];
    sY1[t] = boxes[t * 4 + 1];
    sX2[t] = boxes[t * 4 + 2];
    sY2[t] = boxes[t * 4 + 3];
    sArea[t] = (float)inter[t * NDET + t];
    __syncthreads();

    const float si = sScore[t];
    int rank = 0;
    for (int j = 0; j < NDET; ++j) {
        float sj = sScore[j];
        rank += (sj > si || (sj == si && j < t)) ? 1 : 0;
    }
    sOrder[rank] = t;
    __syncthreads();

    const int oT = sOrder[t];
    sKeepS[t] = (sArea[oT] > 20.0f) ? 1 : 0;
    __syncthreads();

    for (int i = 0; i < NDET - 1; ++i) {
        __syncthreads();
        int kill = 0;
        if (t > i && sKeepS[t] && sKeepS[i]) {
            int   oi   = sOrder[i];
            float ov   = (float)inter[oi * NDET + oT];
            float ai   = sArea[oi], aj = sArea[oT];
            float amin = fminf(ai, aj), amax = fmaxf(ai, aj);
            float iomin = ov / fmaxf(amin, 1.0f);
            float iomax = ov / fmaxf(amax, 1.0f);
            kill = (iomin > 0.9f || iomax > 0.5f) ? 1 : 0;
        }
        __syncthreads();
        if (kill) sKeepS[t] = 0;
    }
    __syncthreads();
    sKeepO[oT] = sKeepS[t];
    __syncthreads();

    const int keepLbl = sKeepO[t] && (sLabel[t] == 1);
    const int high    = (si >= 0.5f) ? 1 : 0;
    __syncthreads();
    sKeepO[t] = keepLbl;
    __syncthreads();

    const float x1j = sX1[t], y1j = sY1[t], x2j = sX2[t], y2j = sY2[t];
    const float barea = fmaxf((x2j - x1j) * (y2j - y1j), 1e-6f);
    float cover = 0.0f;
    for (int i = 0; i < NDET; ++i) {
        if (i == t) continue;
        if (!(sKeepO[i] && sScore[i] >= 0.5f)) continue;
        float ix1 = fmaxf(sX1[i], x1j), iy1 = fmaxf(sY1[i], y1j);
        float ix2 = fminf(sX2[i], x2j), iy2 = fminf(sY2[i], y2j);
        float bint = fmaxf(ix2 - ix1, 0.0f) * fmaxf(iy2 - iy1, 0.0f);
        cover = fmaxf(cover, bint / barea);
    }
    const float vis = 1.0f - cover;
    const int finalKeep = keepLbl && (high || vis < 0.1f);
    out[t] = si * (finalKeep ? 1.0f : 0.0f);
}

extern "C" void kernel_launch(void* const* d_in, const int* in_sizes, int n_in,
                              void* d_out, int out_size, void* d_ws, size_t ws_size,
                              hipStream_t stream) {
    const float* boxes  = (const float*)d_in[0];
    const float* scores = (const float*)d_in[1];
    const int*   labels = (const int*)  d_in[2];
    const float* masks  = (const float*)d_in[3];
    float* out  = (float*)d_out;
    int*  inter = (int*)d_ws;

    pf_zero_kernel<<<(NDET * NDET + 255) / 256, 256, 0, stream>>>(inter, NDET * NDET);

    const size_t ldsBytes = (size_t)NDET * STRIDE;
    pf_gram_kernel<<<256, 256, ldsBytes, stream>>>(masks, inter);

    pf_epilogue_kernel<<<1, NDET, 0, stream>>>(boxes, scores, labels, inter, out);
}
